// CoAttLayer_86414741995961
// MI455X (gfx1250) — matrix operand order repaired and checked by reading, not compiled, not run on hardware
//
#include <hip/hip_runtime.h>

// CoAttention layer, fused for MI455X (gfx1250, wave32, bf16 WMMA).
// Shapes: B=64, T=N=1024, d=64, k=128.
// Workspace layout (fp32): RW[64,1024,64] | ArP[64,128,1024] | Pp[64,128,1024]
//                          | sp[64,1024] | sr[64,1024]  (~81 MB total)

typedef __attribute__((ext_vector_type(16))) __bf16 v16bf;
typedef __attribute__((ext_vector_type(8)))  __bf16 v8bf;
typedef __attribute__((ext_vector_type(8)))  float  v8f;

#if __has_builtin(__builtin_amdgcn_tanhf)
#define TANH_F32(x) __builtin_amdgcn_tanhf(x)      // 1 HW trans op (v_tanh_f32)
#elif __has_builtin(__builtin_amdgcn_tanh_f32)
#define TANH_F32(x) __builtin_amdgcn_tanh_f32(x)
#else
#define TANH_F32(x) tanhf(x)
#endif

// A fragment (16-bit, 16x32, [M][K] row-major LDS tile with row stride 64):
// lane lo -> M row; VGPR0-3 hold K = hi*8..hi*8+7, VGPR4-7 hold K = 16+hi*8..
__device__ __forceinline__ v16bf frag_a(const __bf16* base, int row, int koff, int hi) {
  v16bf a;
  uint4* p = reinterpret_cast<uint4*>(&a);
  const __bf16* s = base + row * 64 + koff + hi * 8;
  p[0] = *reinterpret_cast<const uint4*>(s);
  p[1] = *reinterpret_cast<const uint4*>(s + 16);
  return a;
}

// B fragment (16-bit, 32x16, stored transposed as [N][K], row stride 64):
// lane lo -> N row; VGPR0-3 hold K = hi*8..hi*8+7, VGPR4-7 hold K = 16+hi*8..
__device__ __forceinline__ v16bf frag_b(const __bf16* base, int row, int koff, int hi) {
  v16bf b;
  uint4* p = reinterpret_cast<uint4*>(&b);
  const __bf16* s = base + row * 64 + koff + hi * 8;
  p[0] = *reinterpret_cast<const uint4*>(s);
  p[1] = *reinterpret_cast<const uint4*>(s + 16);
  return b;
}

// Convert float4 -> 4x bf16, one 8-byte store.
__device__ __forceinline__ void cvt_store4(__bf16* dst, const float4 v) {
  union { __bf16 h[4]; uint2 u; } pk;
  pk.h[0] = (__bf16)v.x;
  pk.h[1] = (__bf16)v.y;
  pk.h[2] = (__bf16)v.z;
  pk.h[3] = (__bf16)v.w;
  *reinterpret_cast<uint2*>(dst) = pk.u;
}

// -------------------- Stage A: small precompute kernels --------------------

// RW[b,t,:] = review[b,t,:] @ Wl   (Wl is [d][D] row-major)
__global__ __launch_bounds__(256) void coatt_rw_kernel(
    const float* __restrict__ review, const float* __restrict__ Wl,
    float* __restrict__ RW) {
  __shared__ float sW[64 * 64];
  __shared__ float sR[16 * 64];
  const int tid = threadIdx.x;
  const int b = blockIdx.y;
  const int t0 = blockIdx.x * 16;
  for (int i = tid; i < 64 * 64; i += 256) sW[i] = Wl[i];
  for (int i = tid; i < 16 * 64; i += 256)
    sR[i] = review[((size_t)b * 1024 + t0) * 64 + i];
  __syncthreads();
  for (int i = tid; i < 16 * 64; i += 256) {
    int tr = i >> 6, dcol = i & 63;
    float acc = 0.0f;
    for (int dd = 0; dd < 64; ++dd) acc += sR[tr * 64 + dd] * sW[dd * 64 + dcol];
    RW[((size_t)b * 1024 + t0 + tr) * 64 + dcol] = acc;
  }
}

// Out[b,k,t] = sum_d Wm[k,d] * X[b,t,d]   (Wm is [128][64])
__global__ __launch_bounds__(256) void coatt_proj_kernel(
    const float* __restrict__ Wm, const float* __restrict__ X,
    float* __restrict__ Out) {
  __shared__ float sW[128 * 64];
  __shared__ float sX[64 * 64];
  const int tid = threadIdx.x;
  const int b = blockIdx.y;
  const int t0 = blockIdx.x * 64;
  for (int i = tid; i < 128 * 64; i += 256) sW[i] = Wm[i];
  for (int i = tid; i < 64 * 64; i += 256)
    sX[i] = X[((size_t)b * 1024 + t0) * 64 + i];
  __syncthreads();
  for (int i = tid; i < 128 * 64; i += 256) {
    int k = i >> 6, tr = i & 63;
    float acc = 0.0f;
    for (int dd = 0; dd < 64; ++dd) acc += sW[k * 64 + dd] * sX[tr * 64 + dd];
    Out[((size_t)b * 128 + k) * 1024 + t0 + tr] = acc;
  }
}

// -------------------- Main fused kernel (both heavy paths) --------------------
// s[b, jbase+j] = sum_k wh[k] * tanh(Bias[b,k,j] + sum_c W[b,k,c]*tanh(U[b,c]·V[b,j]))
// One workgroup = (batch b, 128 j-columns). 256 threads = 8 waves.
__global__ __launch_bounds__(256) void coatt_gemm_kernel(
    const float* __restrict__ U, const float* __restrict__ V,
    const float* __restrict__ W, const float* __restrict__ Bias,
    const float* __restrict__ wh, float* __restrict__ sOut) {
  __shared__ __align__(16) __bf16 Ult[64 * 64];    // U tile  [c=64][d=64]
  __shared__ __align__(16) __bf16 Vlt[128 * 64];   // V tile  [j=128][d=64]
  __shared__ __align__(16) __bf16 Wlt[128 * 64];   // W tile  [k=128][c=64]
  __shared__ __align__(16) __bf16 Llt[128 * 64];   // L tile  [j=128][c=64] (B-operand layout)
  __shared__ float sbuf[128];

  const int tid = threadIdx.x;
  const int b = blockIdx.y;
  const int jbase = blockIdx.x * 128;
  const int wave = tid >> 5;
  const int lane = tid & 31;
  const int lo = lane & 15;
  const int hi = lane >> 4;

  if (tid < 128) sbuf[tid] = 0.0f;

  // Stage V tile once (j fixed for this workgroup): wide b128 loads, bf16 LDS.
  {
    const float4* src = reinterpret_cast<const float4*>(V + ((size_t)b * 1024 + jbase) * 64);
    for (int i4 = tid; i4 < 128 * 16; i4 += 256) cvt_store4(Vlt + i4 * 4, src[i4]);
  }

  const int kt0 = (wave & 3) * 2;   // 2 k-tiles (32 rows) per wave
  const int jt0 = (wave >> 2) * 4;  // 4 j-tiles (64 cols) per wave

  v8f acc[2][4];
  for (int a = 0; a < 2; ++a)
    for (int c = 0; c < 4; ++c)
      for (int r = 0; r < 8; ++r) acc[a][c][r] = 0.0f;

  for (int cbase = 0; cbase < 1024; cbase += 64) {
    // Stage U tile (linear copy) and W tile (strided rows), fp32 -> bf16, b128 loads.
    {
      const float4* src = reinterpret_cast<const float4*>(U + ((size_t)b * 1024 + cbase) * 64);
      for (int i4 = tid; i4 < 64 * 16; i4 += 256) cvt_store4(Ult + i4 * 4, src[i4]);
    }
    for (int i4 = tid; i4 < 128 * 16; i4 += 256) {
      int k = i4 >> 4, c4 = (i4 & 15) * 4;
      float4 v = *reinterpret_cast<const float4*>(
          W + ((size_t)b * 128 + k) * 1024 + cbase + c4);
      cvt_store4(Wlt + k * 64 + c4, v);
    }
    __syncthreads();

    // GEMM1: L[c, 16*wave + :16] = tanh(U_tile @ V_tileᵀ), K = d = 64
    {
      const v16bf bv0 = frag_b(Vlt, wave * 16 + lo, 0, hi);   // invariant across ct
      const v16bf bv1 = frag_b(Vlt, wave * 16 + lo, 32, hi);
      for (int ct = 0; ct < 4; ++ct) {
        v8f c1 = {0.f, 0.f, 0.f, 0.f, 0.f, 0.f, 0.f, 0.f};
        c1 = __builtin_amdgcn_wmma_f32_16x16x32_bf16(false, frag_a(Ult, ct * 16 + lo, 0, hi),
                                                     false, bv0, (short)0, c1, false, false);
        c1 = __builtin_amdgcn_wmma_f32_16x16x32_bf16(false, frag_a(Ult, ct * 16 + lo, 32, hi),
                                                     false, bv1, (short)0, c1, false, false);
        // C layout: N=lo (j), M=r+hi*8 (c). tanh, cvt bf16, one 16B LDS store
        // lands directly in the [j][c] layout GEMM2's B fragment wants.
        v8bf h;
        for (int r = 0; r < 8; ++r) h[r] = (__bf16)TANH_F32(c1[r]);
        *reinterpret_cast<uint4*>(&Llt[(wave * 16 + lo) * 64 + ct * 16 + hi * 8]) =
            *reinterpret_cast<uint4*>(&h);
      }
    }
    __syncthreads();

    // GEMM2: acc[k,j] += W_tile @ L_tile, K = c = 64
    for (int cc = 0; cc < 2; ++cc) {
      v16bf afr[2], bfr[4];
      for (int kt = 0; kt < 2; ++kt)
        afr[kt] = frag_a(Wlt, (kt0 + kt) * 16 + lo, cc * 32, hi);
      for (int jj = 0; jj < 4; ++jj)
        bfr[jj] = frag_b(Llt, (jt0 + jj) * 16 + lo, cc * 32, hi);
      for (int kt = 0; kt < 2; ++kt)
        for (int jj = 0; jj < 4; ++jj)
          acc[kt][jj] = __builtin_amdgcn_wmma_f32_16x16x32_bf16(
              false, afr[kt], false, bfr[jj], (short)0, acc[kt][jj], false, false);
    }
    __syncthreads();
  }

  // Epilogue: H = tanh(Bias + acc); s[j] += wh[k]*H[k,j] reduced over k
  for (int kt = 0; kt < 2; ++kt) {
    const int kb = (kt0 + kt) * 16 + hi * 8;
    for (int jj = 0; jj < 4; ++jj) {
      const int jloc = (jt0 + jj) * 16 + lo;
      float part = 0.0f;
      for (int r = 0; r < 8; ++r) {
        const int k = kb + r;
        float hv = TANH_F32(Bias[((size_t)b * 128 + k) * 1024 + jbase + jloc] +
                            acc[kt][jj][r]);
        part += wh[k] * hv;
      }
      atomicAdd(&sbuf[jloc], part);
    }
  }
  __syncthreads();
  if (tid < 128) sOut[(size_t)b * 1024 + jbase + tid] = sbuf[tid];
}

// -------------------- Finalize: softmax + weighted feature sums --------------------
__global__ __launch_bounds__(256) void coatt_finalize_kernel(
    const float* __restrict__ sp, const float* __restrict__ sr,
    const float* __restrict__ post, const float* __restrict__ review,
    float* __restrict__ out) {
  __shared__ float red[256];
  __shared__ float wgt[1024];
  __shared__ float fpart[4][64];
  const int tid = threadIdx.x;
  const int b = blockIdx.x;

  for (int which = 0; which < 2; ++which) {
    const float* s = (which == 0) ? sp : sr;
    const float* feat = (which == 0) ? post : review;

    float m = -3.4e38f;
    for (int i = tid; i < 1024; i += 256) m = fmaxf(m, s[(size_t)b * 1024 + i]);
    red[tid] = m;
    __syncthreads();
    for (int o = 128; o > 0; o >>= 1) {
      if (tid < o) red[tid] = fmaxf(red[tid], red[tid + o]);
      __syncthreads();
    }
    m = red[0];
    __syncthreads();

    float sum = 0.0f;
    for (int i = tid; i < 1024; i += 256) {
      float e = __expf(s[(size_t)b * 1024 + i] - m);
      wgt[i] = e;
      sum += e;
    }
    red[tid] = sum;
    __syncthreads();
    for (int o = 128; o > 0; o >>= 1) {
      if (tid < o) red[tid] += red[tid + o];
      __syncthreads();
    }
    const float inv = 1.0f / red[0];
    __syncthreads();

    const int dc = tid & 63, chunk = tid >> 6;
    float acc = 0.0f;
    for (int i = chunk * 256; i < chunk * 256 + 256; ++i)
      acc += feat[((size_t)b * 1024 + i) * 64 + dc] * wgt[i];
    fpart[chunk][dc] = acc * inv;
    __syncthreads();
    if (tid < 64)
      out[b * 128 + which * 64 + tid] =
          fpart[0][tid] + fpart[1][tid] + fpart[2][tid] + fpart[3][tid];
    __syncthreads();
  }
}

// -------------------- Launch --------------------
extern "C" void kernel_launch(void* const* d_in, const int* in_sizes, int n_in,
                              void* d_out, int out_size, void* d_ws, size_t ws_size,
                              hipStream_t stream) {
  const float* review = (const float*)d_in[0];  // [64,1024,64]
  const float* post   = (const float*)d_in[1];  // [64,1024,64]
  const float* Wl     = (const float*)d_in[2];  // [64,64]
  const float* Wr     = (const float*)d_in[3];  // [128,64]
  const float* Wp     = (const float*)d_in[4];  // [128,64]
  const float* whr    = (const float*)d_in[5];  // [1,128]
  const float* whp    = (const float*)d_in[6];  // [1,128]
  float* out = (float*)d_out;                   // [64,128]

  float* ws  = (float*)d_ws;
  float* RW  = ws;                  // 64*1024*64   = 4,194,304 f32
  float* ArP = RW + 4194304;        // 64*128*1024  = 8,388,608 f32
  float* Pp  = ArP + 8388608;       // 64*128*1024  = 8,388,608 f32
  float* sp  = Pp + 8388608;        // 64*1024      = 65,536 f32
  float* sr  = sp + 65536;          // 64*1024      = 65,536 f32

  // Stage A: RW = review@Wl ; Ar_pre = Wr@reviewᵀ ; P = Wp@postᵀ
  coatt_rw_kernel<<<dim3(64, 64), 256, 0, stream>>>(review, Wl, RW);
  coatt_proj_kernel<<<dim3(16, 64), 256, 0, stream>>>(Wr, review, ArP);
  coatt_proj_kernel<<<dim3(16, 64), 256, 0, stream>>>(Wp, post, Pp);

  // Path P: s_p[n] = sum_k whp[k] tanh(P + Ar_pre @ L)
  coatt_gemm_kernel<<<dim3(8, 64), 256, 0, stream>>>(RW, post, ArP, Pp, whp, sp);
  // Path R: s_r[t] = sum_k whr[k] tanh(Ar_pre + P @ Lᵀ)
  coatt_gemm_kernel<<<dim3(8, 64), 256, 0, stream>>>(post, RW, Pp, ArP, whr, sr);

  // Softmax + weighted sums -> [co_p | co_r]
  coatt_finalize_kernel<<<64, 256, 0, stream>>>(sp, sr, post, review, out);
}
